// CBP_66718021976583
// MI455X (gfx1250) — compile-verified, matrix operand order checked by reading, not run on hardware
//
#include <hip/hip_runtime.h>
#include <hip/hip_bf16.h>
#include <math.h>

#define BATCH 16
#define CDIM  512
#define NSP   196      // 14*14 spatial positions (K of the Gram GEMM)
#define PDIM  8192
#define PMASK 8191
#define TILEROWS 64    // rows per staged tile
#define TILEELTS (TILEROWS * NSP)   // 12544 floats, contiguous in memory

#define USE_TDM 1      // 1: TENSOR_LOAD_TO_LDS (clang-23 6-arg builtin); 0: cooperative copy

typedef __attribute__((ext_vector_type(2)))  float    v2f;
typedef __attribute__((ext_vector_type(8)))  float    v8f;
typedef __attribute__((ext_vector_type(16))) _Float16 v16h;
typedef __attribute__((ext_vector_type(4)))  unsigned u32x4;
typedef __attribute__((ext_vector_type(8)))  int      i32x8;
typedef __attribute__((ext_vector_type(4)))  int      i32x4;

#if __has_builtin(__builtin_amdgcn_wmma_f32_16x16x4_f32)
#define HAVE_F32WMMA 1
#else
#define HAVE_F32WMMA 0
#endif

// ---------------------------------------------------------------------------
// Kernel A: zero the per-batch accumulator y[16][8192] in workspace.
// ---------------------------------------------------------------------------
__global__ void cbp_zero(float* __restrict__ p, int n) {
    int i = blockIdx.x * blockDim.x + threadIdx.x;
    if (i < n) p[i] = 0.0f;
}

#if USE_TDM
// Issue a TDM 2D tile load: global (rows x row_len f32, row stride = stride
// elements) -> LDS at byte offset lds_off. Descriptor layout per CDNA5 ISA
// ch.8 (D# group0/group1). Caller must be a single wave; completion via
// s_wait_tensorcnt.
__device__ __forceinline__ void tdm_load_2d(unsigned lds_off, const float* gsrc,
                                            unsigned rows, unsigned row_len,
                                            unsigned stride) {
    unsigned long long ga = (unsigned long long)(const void*)gsrc;
    u32x4 g0;
    g0[0] = 1u;                                   // count=1 (valid), user mode
    g0[1] = lds_off;                              // lds_addr (bytes)
    g0[2] = (unsigned)(ga & 0xFFFFFFFFu);         // global_addr[31:0]
    g0[3] = (unsigned)((ga >> 32) & 0x1FFFFFFu)   // global_addr[56:32]
          | (2u << 30);                           // type=2 ("image")
    i32x8 g1;
    g1[0] = (int)(2u << 16);                      // wg_mask=0, data_size=2 (4B)
    g1[1] = (int)((row_len & 0xFFFFu) << 16);     // tensor_dim0[15:0] @ bits 63:48
    g1[2] = (int)((row_len >> 16) & 0xFFFFu)      // tensor_dim0[31:16]
          | (int)((rows & 0xFFFFu) << 16);        // tensor_dim1[15:0]
    g1[3] = (int)((rows >> 16) & 0xFFFFu)         // tensor_dim1[31:16]
          | (int)((row_len & 0xFFFFu) << 16);     // tile_dim0
    g1[4] = (int)(rows & 0xFFFFu);                // tile_dim1 (tile_dim2 = 0)
    g1[5] = (int)stride;                          // tensor_dim0_stride[31:0]
    g1[6] = 0;                                    // stride[47:32], dim1_stride lo
    g1[7] = 0;
    i32x4 z4 = {0, 0, 0, 0};                      // groups 2/3 unused (2D tensor)
    i32x8 z8 = {0, 0, 0, 0, 0, 0, 0, 0};          // extra group (clang-23 form)
    __builtin_amdgcn_tensor_load_to_lds(g0, g1, z4, z4, z8, 0);
}
#endif

// ---------------------------------------------------------------------------
// Kernel B: per (batch, 64-row c1 stripe) workgroup.
//   - TDM-stages the 64x196 A stripe once; double-buffers the eight 64x196
//     B tiles so the TENSOR_LOAD_TO_LDS for tile j+1 overlaps the WMMA work
//     on tile j (one wave issues, s_wait_tensorcnt + barrier publishes),
//   - computes the 64x512 Gram slab G = X*X^T with v_wmma_f32_16x16x4_f32;
//     each wave shares one A fragment across two subtiles (2 accumulators),
//   - scatters s1[c1]*s2[c2]*G[c1,c2] into an LDS 8192-bin histogram at
//     (h1[c1]+h2[c2]) & 8191 via ds_add_f32,
//   - flushes histogram with global_atomic_add_f32.
// ---------------------------------------------------------------------------
__global__ __launch_bounds__(256) void cbp_gram_scatter(
    const float* __restrict__ x,
    const int*   __restrict__ h1, const int*   __restrict__ h2,
    const float* __restrict__ s1, const float* __restrict__ s2,
    float* __restrict__ yacc_g)
{
    __shared__ float ly[PDIM];           // 32 KB histogram
    __shared__ float lsA[TILEELTS];      // 49 KB A stripe (c1 rows)
    __shared__ float lsB[2][TILEELTS];   // 2x 49 KB B tiles (double buffer)
    __shared__ int   lh1[64];
    __shared__ float ls1[64];
    __shared__ int   lh2[CDIM];
    __shared__ float ls2[CDIM];

    const int tid     = threadIdx.x;
    const int b       = blockIdx.x >> 3;
    const int stripe  = blockIdx.x & 7;
    const int c1_base = stripe * 64;

    for (int p = tid; p < PDIM; p += 256) ly[p] = 0.0f;
    if (tid < 64) { lh1[tid] = h1[c1_base + tid]; ls1[tid] = s1[c1_base + tid]; }
    for (int c = tid; c < CDIM; c += 256) { lh2[c] = h2[c]; ls2[c] = s2[c]; }

    const float* xb = x + (size_t)b * CDIM * NSP;

    // ---- prologue: stage A stripe and B tile 0 ---------------------------
#if USE_TDM
    if (tid < 32) {
        // Low 32 bits of a flat shared pointer are the LDS byte offset.
        tdm_load_2d((unsigned)(unsigned long long)(void*)lsA,
                    xb + (size_t)c1_base * NSP, TILEROWS, NSP, NSP);
        tdm_load_2d((unsigned)(unsigned long long)(void*)lsB[0],
                    xb, TILEROWS, NSP, NSP);
        __builtin_amdgcn_s_wait_tensorcnt(0);
    }
#else
    {
        const float4* sa = (const float4*)(xb + (size_t)c1_base * NSP);
        const float4* sb = (const float4*)xb;
        float4* da = (float4*)lsA;
        float4* db = (float4*)lsB[0];
        for (int i = tid; i < TILEELTS / 4; i += 256) { da[i] = sa[i]; db[i] = sb[i]; }
    }
#endif
    __syncthreads();                     // lsA, lsB[0], tables ready

    const int wave = tid >> 5;
    const int lane = tid & 31;
    const int hl   = lane >> 4;   // lane[4]: selects K sub-pair in A/B frags
    const int mn   = lane & 15;   // A row (M) / B column (N) index

    // Wave -> (i, jj pair): one A fragment feeds two B subtiles.
    const int i   = wave >> 1;           // c1 16-block (0..3)
    const int jj0 = (wave & 1) * 2;      // first c2 16-block of the pair

    const float* arow = lsA + (size_t)(16 * i + mn) * NSP;
    const int l1b = 16 * i + 8 * hl;     // local c1 of acc[0] in D layout

    for (int j = 0; j < 8; ++j) {        // c2 64-wide tiles
        const int cur = j & 1;
        const int nxt = cur ^ 1;
        const int c2_base = j * 64;

        // ---- prefetch next B tile while computing this one ---------------
        if (j < 7) {
#if USE_TDM
            if (tid < 32)
                tdm_load_2d((unsigned)(unsigned long long)(void*)lsB[nxt],
                            xb + (size_t)(j + 1) * 64 * NSP, TILEROWS, NSP, NSP);
#else
            const float4* src = (const float4*)(xb + (size_t)(j + 1) * 64 * NSP);
            float4*       dst = (float4*)lsB[nxt];
            for (int q = tid; q < TILEELTS / 4; q += 256) dst[q] = src[q];
#endif
        }

        const float* b0row = lsB[cur] + (size_t)(16 * jj0       + mn) * NSP;
        const float* b1row = lsB[cur] + (size_t)(16 * (jj0 + 1) + mn) * NSP;

        v8f acc0 = {0.f, 0.f, 0.f, 0.f, 0.f, 0.f, 0.f, 0.f};
        v8f acc1 = {0.f, 0.f, 0.f, 0.f, 0.f, 0.f, 0.f, 0.f};

#if HAVE_F32WMMA
        // fp32 matrix pipe, K=4 per issue: 49 steps cover K=196 exactly.
        // One shared A frag : two B frags : two WMMAs per step.
        for (int k0 = 0; k0 < NSP; k0 += 4) {
            const int ka = k0 + 2 * hl;          // 8-byte aligned (ka even)
            v2f a  = *reinterpret_cast<const v2f*>(arow  + ka);
            v2f b0 = *reinterpret_cast<const v2f*>(b0row + ka);
            v2f b1 = *reinterpret_cast<const v2f*>(b1row + ka);
            acc0 = __builtin_amdgcn_wmma_f32_16x16x4_f32(
                false, a, false, b0, (short)0, acc0, false, false);
            acc1 = __builtin_amdgcn_wmma_f32_16x16x4_f32(
                false, a, false, b1, (short)0, acc1, false, false);
        }
#else
        // fp16 fallback (codegen-confirmed builtin). K padded 196 -> 224.
        for (int k0 = 0; k0 < 224; k0 += 32) {
            v16h a, b0, b1;
#pragma unroll
            for (int v = 0; v < 8; ++v) {
                const int kb = k0 + ((v & 3) << 1) + (hl << 3) + ((v >> 2) << 4);
                const float a0  = (kb     < NSP) ? arow[kb]      : 0.0f;
                const float a1  = (kb + 1 < NSP) ? arow[kb + 1]  : 0.0f;
                const float b00 = (kb     < NSP) ? b0row[kb]     : 0.0f;
                const float b01 = (kb + 1 < NSP) ? b0row[kb + 1] : 0.0f;
                const float b10 = (kb     < NSP) ? b1row[kb]     : 0.0f;
                const float b11 = (kb + 1 < NSP) ? b1row[kb + 1] : 0.0f;
                a[2 * v]  = (_Float16)a0;  a[2 * v + 1]  = (_Float16)a1;
                b0[2 * v] = (_Float16)b00; b0[2 * v + 1] = (_Float16)b01;
                b1[2 * v] = (_Float16)b10; b1[2 * v + 1] = (_Float16)b11;
            }
            acc0 = __builtin_amdgcn_wmma_f32_16x16x32_f16(
                false, a, false, b0, (short)0, acc0, false, false);
            acc1 = __builtin_amdgcn_wmma_f32_16x16x32_f16(
                false, a, false, b1, (short)0, acc1, false, false);
        }
#endif
        // Scatter straight from the D-matrix registers.
        // D layout: VGPR r, lanes 0-15 -> M=r, lanes 16-31 -> M=r+8; N=lane%16.
#pragma unroll
        for (int s = 0; s < 2; ++s) {
            const v8f&  acc = s ? acc1 : acc0;
            const int   c2g = c2_base + 16 * (jj0 + s) + mn;
            const int   hc2 = lh2[c2g];
            const float sc2 = ls2[c2g];
#pragma unroll
            for (int r = 0; r < 8; ++r) {
                const int tgt = (lh1[l1b + r] + hc2) & PMASK;
                atomicAdd(&ly[tgt], ls1[l1b + r] * sc2 * acc[r]);   // ds_add_f32
            }
        }

        if (j < 7) {
#if USE_TDM
            if (tid < 32) __builtin_amdgcn_s_wait_tensorcnt(0);
#endif
            __syncthreads();             // publish lsB[nxt]; retire lsB[cur]
        }
    }

    __syncthreads();
    float* yb = yacc_g + (size_t)b * PDIM;
    for (int p = tid; p < PDIM; p += 256)
        atomicAdd(&yb[p], ly[p]);                              // global_atomic_add_f32
}

// ---------------------------------------------------------------------------
// Kernel C: per-batch signed sqrt + L2 normalize.
// ---------------------------------------------------------------------------
__global__ __launch_bounds__(256) void cbp_finalize(
    const float* __restrict__ yg, float* __restrict__ out)
{
    __shared__ float red[256];
    const int b   = blockIdx.x;
    const int tid = threadIdx.x;
    const float* yb = yg + (size_t)b * PDIM;

    float tv[PDIM / 256];
    float ss = 0.0f;
#pragma unroll
    for (int q = 0; q < PDIM / 256; ++q) {
        const int   p   = tid + q * 256;
        const float v   = yb[p];
        const float sgn = (v > 0.0f) ? 1.0f : ((v < 0.0f) ? -1.0f : 0.0f);
        const float t   = sgn * sqrtf(fabsf(v) + 1e-8f);
        tv[q] = t;
        ss += t * t;
    }
    red[tid] = ss;
    __syncthreads();
    for (int o = 128; o > 0; o >>= 1) {
        if (tid < o) red[tid] += red[tid + o];
        __syncthreads();
    }
    const float scale = 1.0f / fmaxf(sqrtf(red[0]), 1e-12f);
    float* ob = out + (size_t)b * PDIM;
#pragma unroll
    for (int q = 0; q < PDIM / 256; ++q)
        ob[tid + q * 256] = tv[q] * scale;
}

// ---------------------------------------------------------------------------
extern "C" void kernel_launch(void* const* d_in, const int* in_sizes, int n_in,
                              void* d_out, int out_size, void* d_ws, size_t ws_size,
                              hipStream_t stream)
{
    (void)in_sizes; (void)n_in; (void)out_size; (void)ws_size;
    const float* x  = (const float*)d_in[0];  // [16,512,14,14] f32
    const int*   h1 = (const int*)  d_in[1];  // [512]
    const int*   h2 = (const int*)  d_in[2];  // [512]
    const float* s1 = (const float*)d_in[3];  // [512]
    const float* s2 = (const float*)d_in[4];  // [512]

    float* yacc = (float*)d_ws;               // 16*8192 f32 = 512 KB scratch
    const int ny = BATCH * PDIM;

    cbp_zero<<<(ny + 255) / 256, 256, 0, stream>>>(yacc, ny);
    cbp_gram_scatter<<<BATCH * 8, 256, 0, stream>>>(x, h1, h2, s1, s2, yacc);
    cbp_finalize<<<BATCH, 256, 0, stream>>>(yacc, (float*)d_out);
}
